// DeformableConv2d_10307921510980
// MI455X (gfx1250) — compile-verified, matrix operand order checked by reading, not run on hardware
//
#include <hip/hip_runtime.h>
#include <hip/hip_fp16.h>

typedef __attribute__((ext_vector_type(16))) _Float16 v16h;
typedef __attribute__((ext_vector_type(8)))  _Float16 v8h;
typedef __attribute__((ext_vector_type(4)))  _Float16 v4h;
typedef __attribute__((ext_vector_type(8)))  float    v8f;
typedef __attribute__((ext_vector_type(4)))  float    v4f;

#define Bn    8
#define Cc    64
#define Hh    128
#define Ww    128
#define HW    (Hh * Ww)
#define KK    9
#define OffC  18
#define Cout  64
#define Kdim  (Cc * KK)   // 576
#define MT    64          // output positions per block (M tile)
#define KCH   (Kdim / 32) // 18 WMMA K-chunks

// ---------------------------------------------------------------------------
// Stage 0a: NCHW -> NHWC transpose so every bilinear corner and conv tap reads
// 64 contiguous channels (256 B) instead of 64 loads at 64 KB stride.
// One block per (b, h): tile [C=64][W=128] staged through LDS.
// ---------------------------------------------------------------------------
__global__ void transpose_nchw_nhwc(const float* __restrict__ x,
                                    float* __restrict__ xT) {
    __shared__ float tile[Cc][Ww + 1];   // +1 pad kills bank conflicts
    const int b = blockIdx.x / Hh;
    const int h = blockIdx.x % Hh;
    const float* src = x + (size_t)b * Cc * HW + (size_t)h * Ww;
    for (int i = threadIdx.x; i < Cc * Ww; i += blockDim.x) {
        int c = i / Ww, w = i % Ww;                 // coalesced along w
        tile[c][w] = src[(size_t)c * HW + w];
    }
    __syncthreads();
    float* dst = xT + ((size_t)b * HW + (size_t)h * Ww) * Cc;
    for (int i = threadIdx.x; i < Cc * Ww; i += blockDim.x) {
        int w = i / Cc, c = i % Cc;                 // coalesced along c
        dst[(size_t)w * Cc + c] = tile[c][w];
    }
}

// ---------------------------------------------------------------------------
// Stage 0b: one-shot weight prep: f32 [O][C][KK] -> f16 [O][kk*Cc + c].
// 72 KB result stays L2-resident; deform blocks then load it contiguously.
// ---------------------------------------------------------------------------
__global__ void prep_weights(const float* __restrict__ w_def,
                             _Float16* __restrict__ wsh) {
    int i = blockIdx.x * blockDim.x + threadIdx.x;
    if (i >= Cout * Kdim) return;
    int o  = i / Kdim;
    int k  = i % Kdim;
    int kk = k / Cc;
    int c  = k % Cc;
    wsh[i] = (_Float16)w_def[(size_t)o * Kdim + c * KK + kk];
}

// ---------------------------------------------------------------------------
// Stage 1: standard 3x3 conv (64 -> 18 ch) on NHWC input, fp32.
// ---------------------------------------------------------------------------
__global__ void offset_conv_kernel(const float* __restrict__ xT,
                                   const float* __restrict__ w_off,
                                   const float* __restrict__ b_off,
                                   float* __restrict__ offs) {
    int idx = blockIdx.x * blockDim.x + threadIdx.x;
    const int total = Bn * OffC * HW;
    if (idx >= total) return;
    int w  = idx % Ww;
    int h  = (idx / Ww) % Hh;
    int co = (idx / HW) % OffC;
    int b  = idx / (HW * OffC);

    float acc = b_off[co];
#pragma unroll
    for (int ky = 0; ky < 3; ++ky) {
        int yy = h + ky - 1;
        if (yy < 0 || yy >= Hh) continue;
#pragma unroll
        for (int kx = 0; kx < 3; ++kx) {
            int xx = w + kx - 1;
            if (xx < 0 || xx >= Ww) continue;
            const float* p  = xT + ((size_t)b * HW + yy * Ww + xx) * Cc;
            const float* wr = w_off + (size_t)co * Kdim + ky * 3 + kx;
            for (int c = 0; c < Cc; c += 4) {
                v4f v = *(const v4f*)(p + c);
                acc = fmaf(v.x, wr[(c    ) * KK], acc);
                acc = fmaf(v.y, wr[(c + 1) * KK], acc);
                acc = fmaf(v.z, wr[(c + 2) * KK], acc);
                acc = fmaf(v.w, wr[(c + 3) * KK], acc);
            }
        }
    }
    offs[idx] = acc;
}

// Build a v16h A/B fragment from two contiguous 8-half LDS chunks.
// ISA 16-bit fragment layout: lanes 0-15 hold K {kb..kb+7, kb+16..kb+23},
// lanes 16-31 hold K {kb+8..kb+15, kb+24..kb+31}; khalf = 0 or 8.
__device__ __forceinline__ v16h mkfrag(const _Float16* p, int khalf) {
    v8h lo = *(const v8h*)(p + khalf);
    v8h hi = *(const v8h*)(p + 16 + khalf);
    v16h r;
#pragma unroll
    for (int i = 0; i < 8; ++i) { r[i] = lo[i]; r[i + 8] = hi[i]; }
    return r;
}

// ---------------------------------------------------------------------------
// Stage 2: bilinear gather (NHWC, v4f corners) into an f16 A-panel in LDS,
// then WMMA GEMM:  out[M=64 pos][N=64 ch] += A[64][576] * B[576][64].
// K dimension ordered k = kk*64 + c (A writes contiguous; B pre-permuted).
// ---------------------------------------------------------------------------
__global__ void deform_wmma_kernel(const float* __restrict__ xT,
                                   const float* __restrict__ offs,
                                   const _Float16* __restrict__ wsh,
                                   float* __restrict__ out) {
    extern __shared__ _Float16 smem[];
    _Float16* Ash = smem;                       // [MT][Kdim]   (72 KB)
    _Float16* Bsh = smem + (size_t)MT * Kdim;   // [Cout][Kdim] (72 KB)

    const int tid   = threadIdx.x;
    const int baseM = blockIdx.x * MT;

    // ---- weights: contiguous f16 copy (18x b128 per thread, L2-resident) --
    for (int i = tid; i < (Cout * Kdim) / 8; i += blockDim.x)
        *((v8h*)Bsh + i) = *((const v8h*)wsh + i);

    // ---- bilinear sampling: one (pos, kk) pair per iteration ----
    for (int pair = tid; pair < MT * KK; pair += blockDim.x) {
        int pos  = pair / KK;
        int kk   = pair % KK;
        int gpos = baseM + pos;
        int b    = gpos / HW;
        int hw   = gpos % HW;
        int h    = hw / Ww;
        int w    = hw % Ww;
        int ky   = kk / 3;
        int kx   = kk % 3;

        float dy = offs[((size_t)(b * OffC + 2 * kk    )) * HW + hw];
        float dx = offs[((size_t)(b * OffC + 2 * kk + 1)) * HW + hw];
        float py = (float)(h - 1 + ky) + dy;
        float px = (float)(w - 1 + kx) + dx;
        float y0f = floorf(py), x0f = floorf(px);
        float wy = py - y0f, wx = px - x0f;
        int y0 = (int)y0f, x0 = (int)x0f;
        int y1 = y0 + 1,   x1 = x0 + 1;

        float vy0 = (y0 >= 0 && y0 < Hh) ? 1.f : 0.f;
        float vy1 = (y1 >= 0 && y1 < Hh) ? 1.f : 0.f;
        float vx0 = (x0 >= 0 && x0 < Ww) ? 1.f : 0.f;
        float vx1 = (x1 >= 0 && x1 < Ww) ? 1.f : 0.f;
        float w00 = (1.f - wy) * (1.f - wx) * vy0 * vx0;
        float w01 = (1.f - wy) * wx         * vy0 * vx1;
        float w10 = wy * (1.f - wx)         * vy1 * vx0;
        float w11 = wy * wx                 * vy1 * vx1;

        int cy0 = min(max(y0, 0), Hh - 1), cy1 = min(max(y1, 0), Hh - 1);
        int cx0 = min(max(x0, 0), Ww - 1), cx1 = min(max(x1, 0), Ww - 1);

        const float* bbase = xT + (size_t)b * HW * Cc;
        const float* p00 = bbase + (size_t)(cy0 * Ww + cx0) * Cc;
        const float* p01 = bbase + (size_t)(cy0 * Ww + cx1) * Cc;
        const float* p10 = bbase + (size_t)(cy1 * Ww + cx0) * Cc;
        const float* p11 = bbase + (size_t)(cy1 * Ww + cx1) * Cc;

        _Float16* arow = Ash + (size_t)pos * Kdim + kk * Cc;  // contiguous
        for (int c = 0; c < Cc; c += 4) {
            v4f a0 = *(const v4f*)(p00 + c);
            v4f a1 = *(const v4f*)(p01 + c);
            v4f a2 = *(const v4f*)(p10 + c);
            v4f a3 = *(const v4f*)(p11 + c);
            v4h r;
            r[0] = (_Float16)(w00 * a0.x + w01 * a1.x + w10 * a2.x + w11 * a3.x);
            r[1] = (_Float16)(w00 * a0.y + w01 * a1.y + w10 * a2.y + w11 * a3.y);
            r[2] = (_Float16)(w00 * a0.z + w01 * a1.z + w10 * a2.z + w11 * a3.z);
            r[3] = (_Float16)(w00 * a0.w + w01 * a1.w + w10 * a2.w + w11 * a3.w);
            *(v4h*)(arow + c) = r;
        }
    }
    __syncthreads();

    // ---- WMMA GEMM: 8 waves cover 4x4 grid of 16x16 tiles (2 tiles/wave) --
    const int lane  = tid & 31;
    const int wv    = tid >> 5;       // 0..7
    const int tn    = wv & 3;         // N tile (B fragment shared by both tiles)
    const int tm0   = wv >> 2;        // M tiles tm0 and tm0+2
    const int mrow  = lane & 15;
    const int khalf = (lane < 16) ? 0 : 8;

    const _Float16* Arow0 = Ash + (size_t)(tm0 * 16 + mrow) * Kdim;
    const _Float16* Arow1 = Ash + (size_t)((tm0 + 2) * 16 + mrow) * Kdim;
    const _Float16* Brow  = Bsh + (size_t)(tn * 16 + mrow) * Kdim;

    v8f acc0 = {};
    v8f acc1 = {};
#pragma unroll 2
    for (int kc = 0; kc < KCH; ++kc) {
        int kb = kc * 32;
        v16h bf = mkfrag(Brow  + kb, khalf);
        v16h a0 = mkfrag(Arow0 + kb, khalf);
        v16h a1 = mkfrag(Arow1 + kb, khalf);
        acc0 = __builtin_amdgcn_wmma_f32_16x16x32_f16(
            false, a0, false, bf, (short)0, acc0, false, false);
        acc1 = __builtin_amdgcn_wmma_f32_16x16x32_f16(
            false, a1, false, bf, (short)0, acc1, false, false);
    }

    // ---- store: lane owns channel o; 8 consecutive positions per acc.
    // Non-temporal: output is write-once, keep L2 for the gather working set.
    const int n     = lane & 15;
    const int o     = tn * 16 + n;
    const int rbase = (lane < 16) ? 0 : 8;
    {
        int gpos = baseM + tm0 * 16 + rbase;   // tile never crosses batch b
        int b    = gpos / HW;
        int hw   = gpos % HW;
        float* op = out + ((size_t)(b * Cout + o)) * HW + hw;
        v4f lo = {acc0[0], acc0[1], acc0[2], acc0[3]};
        v4f hi = {acc0[4], acc0[5], acc0[6], acc0[7]};
        __builtin_nontemporal_store(lo, (v4f*)op);
        __builtin_nontemporal_store(hi, (v4f*)(op + 4));
    }
    {
        int gpos = baseM + (tm0 + 2) * 16 + rbase;
        int b    = gpos / HW;
        int hw   = gpos % HW;
        float* op = out + ((size_t)(b * Cout + o)) * HW + hw;
        v4f lo = {acc1[0], acc1[1], acc1[2], acc1[3]};
        v4f hi = {acc1[4], acc1[5], acc1[6], acc1[7]};
        __builtin_nontemporal_store(lo, (v4f*)op);
        __builtin_nontemporal_store(hi, (v4f*)(op + 4));
    }
}

extern "C" void kernel_launch(void* const* d_in, const int* in_sizes, int n_in,
                              void* d_out, int out_size, void* d_ws, size_t ws_size,
                              hipStream_t stream) {
    const float* x     = (const float*)d_in[0];
    const float* w_off = (const float*)d_in[1];
    const float* b_off = (const float*)d_in[2];
    const float* w_def = (const float*)d_in[3];
    float* out  = (float*)d_out;

    // workspace: xT (NHWC, 33.5 MB) | offsets (9.4 MB) | f16 weights (72 KB)
    float*     xT   = (float*)d_ws;
    float*     offs = xT + (size_t)Bn * Cc * HW;
    _Float16*  wsh  = (_Float16*)(offs + (size_t)Bn * OffC * HW);

    transpose_nchw_nhwc<<<Bn * Hh, 256, 0, stream>>>(x, xT);

    prep_weights<<<(Cout * Kdim + 255) / 256, 256, 0, stream>>>(w_def, wsh);

    const int offTotal = Bn * OffC * HW;
    offset_conv_kernel<<<(offTotal + 255) / 256, 256, 0, stream>>>(
        xT, w_off, b_off, offs);

    const int nblocks = (Bn * HW) / MT;  // 2048
    const size_t lds  = (size_t)(MT * Kdim + Cout * Kdim) * sizeof(_Float16); // 144 KB
    deform_wmma_kernel<<<nblocks, 256, lds, stream>>>(xT, offs, wsh, out);
}
